// CompressedLinear_3367254360324
// MI455X (gfx1250) — matrix operand order repaired and checked by reading, not compiled, not run on hardware
//
#include <hip/hip_runtime.h>

#define IN_F   4096
#define OUT_F  11008
#define MROWS  8192   // BATCH * SEQ

typedef _Float16 v16h __attribute__((ext_vector_type(16)));
typedef _Float16 v8h  __attribute__((ext_vector_type(8)));
typedef _Float16 v4h  __attribute__((ext_vector_type(4)));
typedef float    v8f  __attribute__((ext_vector_type(8)));
typedef float    v4f  __attribute__((ext_vector_type(4)));
typedef int      v4i  __attribute__((ext_vector_type(4)));

// ---------------------------------------------------------------------------
// Prologue 1: x fp32 -> f16  (lossy cast; f16 keeps 11 mantissa bits)
// ---------------------------------------------------------------------------
__global__ void cvt_x_f32_to_f16(const float* __restrict__ x,
                                 _Float16* __restrict__ xh, int n4) {
  int i = blockIdx.x * blockDim.x + threadIdx.x;
  if (i < n4) {
    v4f v = ((const v4f*)x)[i];
    v4h h;
    h[0] = (_Float16)v[0];
    h[1] = (_Float16)v[1];
    h[2] = (_Float16)v[2];
    h[3] = (_Float16)v[3];
    ((v4h*)xh)[i] = h;
  }
}

// ---------------------------------------------------------------------------
// Prologue 2: weight int32 (int8-valued) -> f16 (exact: |w| <= 127 < 2^11)
// ---------------------------------------------------------------------------
__global__ void cvt_w_i32_to_f16(const int* __restrict__ w,
                                 _Float16* __restrict__ wh, int n4) {
  int i = blockIdx.x * blockDim.x + threadIdx.x;
  if (i < n4) {
    v4i v = ((const v4i*)w)[i];
    v4h h;
    h[0] = (_Float16)v[0];
    h[1] = (_Float16)v[1];
    h[2] = (_Float16)v[2];
    h[3] = (_Float16)v[3];
    ((v4h*)wh)[i] = h;
  }
}

// ---------------------------------------------------------------------------
// Main GEMM: out[m,n] = scale * sum_k xh[m,k]*wh[n,k] + bias[n]
// Block tile 128M x 128N, 8 waves (wave32): wave tile 32M x 64N
// => 2x4 v_wmma_f32_16x16x32_f16 accumulators per wave, K step 32.
// ---------------------------------------------------------------------------
__global__ __launch_bounds__(256, 1)
void gemm_f16_wmma(const _Float16* __restrict__ xh,   // [MROWS, IN_F]
                   const _Float16* __restrict__ wh,   // [OUT_F, IN_F]
                   const float* __restrict__ scale_p, // [1]
                   const float* __restrict__ bias,    // [OUT_F]
                   float* __restrict__ out)           // [MROWS, OUT_F]
{
  const int lane = threadIdx.x & 31;
  const int wave = threadIdx.x >> 5;
  const int m16  = lane & 15;   // lane id within half-wave
  const int kg   = lane >> 4;   // half-wave selector (K group)

  const int blockN = blockIdx.x * 128;
  const int blockM = blockIdx.y * 128;
  const int wave_m = blockM + (wave >> 1) * 32;  // 4 waves down M
  const int wave_n = blockN + (wave & 1) * 64;   // 2 waves across N

  // A rows for this lane (contiguous k order, same as B):
  // lane holds A[m, k0 + 16*kg .. +16) (two 8-element loads)
  const _Float16* arow0 = xh + (size_t)(wave_m + m16) * IN_F;
  const _Float16* arow1 = arow0 + (size_t)16 * IN_F;

  // B columns (contiguous k order, matching A): lane holds
  // B[k0 + 16*kg .. +16), n] == wh[n, k0 + 16*kg .. +16)  (contiguous 32B)
  const _Float16* bcol[4];
#pragma unroll
  for (int nt = 0; nt < 4; ++nt)
    bcol[nt] = wh + (size_t)(wave_n + nt * 16 + m16) * IN_F + kg * 16;

  v8f c[2][4];
#pragma unroll
  for (int mt = 0; mt < 2; ++mt)
#pragma unroll
    for (int nt = 0; nt < 4; ++nt)
      c[mt][nt] = (v8f)0.0f;

  for (int k0 = 0; k0 < IN_F; k0 += 32) {
    const int kaLo = k0 + 16 * kg;
    const int kaHi = k0 + 16 * kg + 8;

    v8h lo0 = *(const v8h*)(arow0 + kaLo);
    v8h hi0 = *(const v8h*)(arow0 + kaHi);
    v8h lo1 = *(const v8h*)(arow1 + kaLo);
    v8h hi1 = *(const v8h*)(arow1 + kaHi);

    v16h a[2];
    a[0] = __builtin_shufflevector(lo0, hi0, 0,1,2,3,4,5,6,7,8,9,10,11,12,13,14,15);
    a[1] = __builtin_shufflevector(lo1, hi1, 0,1,2,3,4,5,6,7,8,9,10,11,12,13,14,15);

    v16h b[4];
#pragma unroll
    for (int nt = 0; nt < 4; ++nt)
      b[nt] = *(const v16h*)(bcol[nt] + k0);

#pragma unroll
    for (int mt = 0; mt < 2; ++mt)
#pragma unroll
      for (int nt = 0; nt < 4; ++nt)
        c[mt][nt] = __builtin_amdgcn_wmma_f32_16x16x32_f16(
            /*neg_a=*/false, a[mt], /*neg_b=*/false, b[nt],
            /*c_mod=*/(short)0, c[mt][nt],
            /*reuse_a=*/false, /*reuse_b=*/false);
  }

  // Epilogue: D layout — VGPR i: lanes 0-15 => M=i, N=lane; lanes 16-31 => M=i+8
  const float scale = *scale_p;
#pragma unroll
  for (int mt = 0; mt < 2; ++mt) {
#pragma unroll
    for (int nt = 0; nt < 4; ++nt) {
      const int col = wave_n + nt * 16 + m16;
      const float bv = bias[col];
      const int rbase = wave_m + mt * 16 + 8 * kg;
      float* orow = out + (size_t)rbase * OUT_F + col;
#pragma unroll
      for (int i = 0; i < 8; ++i)
        orow[(size_t)i * OUT_F] = c[mt][nt][i] * scale + bv;
    }
  }
}

// ---------------------------------------------------------------------------
extern "C" void kernel_launch(void* const* d_in, const int* in_sizes, int n_in,
                              void* d_out, int out_size, void* d_ws, size_t ws_size,
                              hipStream_t stream) {
  const float* x      = (const float*)d_in[0];
  const int*   w8     = (const int*)d_in[1];
  const float* scale  = (const float*)d_in[2];
  const float* bias   = (const float*)d_in[3];
  float*       out    = (float*)d_out;

  // Workspace layout: [ xh : MROWS*IN_F f16 | wh : OUT_F*IN_F f16 ]  ~157 MB
  _Float16* xh = (_Float16*)d_ws;
  _Float16* wh = xh + (size_t)MROWS * IN_F;

  {
    int n4 = (MROWS * IN_F) / 4;           // 8,388,608
    cvt_x_f32_to_f16<<<(n4 + 255) / 256, 256, 0, stream>>>(x, xh, n4);
  }
  {
    int n4 = (OUT_F * IN_F) / 4;           // 11,272,192
    cvt_w_i32_to_f16<<<(n4 + 255) / 256, 256, 0, stream>>>(w8, wh, n4);
  }

  dim3 grid(OUT_F / 128, MROWS / 128);     // 86 x 64
  gemm_f16_wmma<<<grid, 256, 0, stream>>>(xh, wh, scale, bias, out);
}
